// STGATBlock_46351287059148
// MI455X (gfx1250) — matrix operand order repaired and checked by reading, not compiled, not run on hardware
//
#include <hip/hip_runtime.h>

typedef __attribute__((ext_vector_type(16))) __bf16 v16bf;
typedef __attribute__((ext_vector_type(8)))  __bf16 v8bf;
typedef __attribute__((ext_vector_type(8)))  float  v8f;
typedef __attribute__((ext_vector_type(4)))  float  v4f;

#define Bq   4
#define Nn   2048
#define Tt   12
#define CIN  2
#define COUT 64
#define TP   10
#define Ff   640
#define Hh   6
#define Oo   16
#define C96  96
#define ROWS 8192   // B*N

// ---------------- Kernel 1: gated temporal conv (TimeBlock) -> x bf16 (8192 x 640)
__global__ __launch_bounds__(64) void k_timeblock(
    const float* __restrict__ X,
    const float* __restrict__ Wt1, const float* __restrict__ bt1,
    const float* __restrict__ Wt2, const float* __restrict__ bt2,
    const float* __restrict__ Wt3, const float* __restrict__ bt3,
    __bf16* __restrict__ Xbf)
{
  int row = blockIdx.x;          // b*N + n
  int o   = threadIdx.x;         // 0..63
  __shared__ float xs[Tt * CIN]; // 24 floats: X[b,n,t,c]
  if (o < Tt * CIN) xs[o] = X[row * (Tt * CIN) + o];
  __syncthreads();

  float w1[6], w2[6], w3[6];
#pragma unroll
  for (int i = 0; i < 6; i++) {       // Wt (Cout,Cin,1,3): idx o*6 + c*3 + k
    w1[i] = Wt1[o * 6 + i];
    w2[i] = Wt2[o * 6 + i];
    w3[i] = Wt3[o * 6 + i];
  }
  float B1 = bt1[o], B2 = bt2[o], B3 = bt3[o];

#pragma unroll
  for (int tp = 0; tp < TP; tp++) {
    float c1 = B1, c2 = B2, c3 = B3;
#pragma unroll
    for (int c = 0; c < CIN; c++)
#pragma unroll
      for (int k = 0; k < 3; k++) {
        float xv = xs[(tp + k) * CIN + c];
        c1 = fmaf(xv, w1[c * 3 + k], c1);
        c2 = fmaf(xv, w2[c * 3 + k], c2);
        c3 = fmaf(xv, w3[c * 3 + k], c3);
      }
    float sg = 1.0f / (1.0f + __expf(-c2));
    float v  = fmaxf(c1 + sg + c3, 0.0f);   // relu(conv1 + sigmoid(conv2) + conv3)
    Xbf[row * Ff + tp * COUT + o] = (__bf16)v;  // F index = t'*Cout + o
  }
}

// ---------------- Kernel 2: pack Wg/Wr (96 x 640) f32 -> bf16
__global__ void k_pack(const float* __restrict__ Wg, const float* __restrict__ Wr,
                       __bf16* __restrict__ Wgbf, __bf16* __restrict__ Wrbf)
{
  int i = blockIdx.x * 256 + threadIdx.x;
  if (i < C96 * Ff) {
    Wgbf[i] = (__bf16)Wg[i];
    Wrbf[i] = (__bf16)Wr[i];
  }
}

// ---------------- Kernel 3: x @ WgT and x @ WrT via bf16 WMMA.
// One wave per block; block handles 16 rows x all 96 cols.
__global__ __launch_bounds__(32) void k_gemm(
    const __bf16* __restrict__ Xbf, const __bf16* __restrict__ Wgbf,
    const __bf16* __restrict__ Wrbf,
    float* __restrict__ Hmat, float* __restrict__ Resmat, __bf16* __restrict__ HbfT)
{
  int i0 = blockIdx.x * 16;
  int l  = threadIdx.x;
  int lm = l & 15;
  bool hi = l >= 16;
  int rowA = i0 + lm;
  const __bf16* xrow = Xbf + rowA * Ff;
  int aoff = hi ? 8 : 0;
  int b = i0 >> 11;            // tile never crosses batch boundary (2048 % 16 == 0)

  // Pull this wave's 1.25 KB A-row toward the WGP before the WMMA chain starts.
#pragma unroll
  for (int pf = 0; pf < Ff; pf += 64)
    __builtin_prefetch((const void*)(xrow + pf), 0, 3);

  for (int cc = 0; cc < 6; cc++) {
    int col = cc * 16 + lm;
    const __bf16* wg = Wgbf + col * Ff + (hi ? 16 : 0);
    const __bf16* wr = Wrbf + col * Ff + (hi ? 16 : 0);
    v8f ch = {};
    v8f cr = {};
    for (int kt = 0; kt < Ff / 32; kt++) {
      int kb = kt * 32;
      // A fragment: lane holds row lm, K = kb + (hi?16:0) + 0..15
      v8bf alo = *(const v8bf*)(xrow + kb + 2 * aoff);
      v8bf ahi = *(const v8bf*)(xrow + kb + 2 * aoff + 8);
      v16bf A;
#pragma unroll
      for (int i = 0; i < 8; i++) { A[i] = alo[i]; A[i + 8] = ahi[i]; }
      // B fragment: lane holds col lm, K = kb + (hi?16:0) + 0..15
      v8bf g0 = *(const v8bf*)(wg + kb);
      v8bf g1 = *(const v8bf*)(wg + kb + 8);
      v16bf Bg;
#pragma unroll
      for (int i = 0; i < 8; i++) { Bg[i] = g0[i]; Bg[i + 8] = g1[i]; }
      ch = __builtin_amdgcn_wmma_f32_16x16x32_bf16(false, A, false, Bg, (short)0, ch, false, false);

      v8bf r0 = *(const v8bf*)(wr + kb);
      v8bf r1 = *(const v8bf*)(wr + kb + 8);
      v16bf Br;
#pragma unroll
      for (int i = 0; i < 8; i++) { Br[i] = r0[i]; Br[i + 8] = r1[i]; }
      cr = __builtin_amdgcn_wmma_f32_16x16x32_bf16(false, A, false, Br, (short)0, cr, false, false);
    }
    // C layout: VGPR r holds row r (lanes 0-15) / row 8+r (lanes 16-31), col = lm
#pragma unroll
    for (int r = 0; r < 8; r++) {
      int rowm = r + (hi ? 8 : 0);
      int grow = i0 + rowm;
      int gcol = cc * 16 + lm;
      Hmat[grow * C96 + gcol]   = ch[r];
      Resmat[grow * C96 + gcol] = cr[r];
      int n = grow & (Nn - 1);
      HbfT[(b * C96 + gcol) * Nn + n] = (__bf16)ch[r];   // per-head transposed bf16 copy
    }
  }
}

// ---------------- Kernel 4: attention pre-logits f1/f2
__global__ void k_f1f2(const float* __restrict__ Hmat,
                       const float* __restrict__ a1, const float* __restrict__ b1,
                       const float* __restrict__ a2, const float* __restrict__ b2,
                       float* __restrict__ f1, float* __restrict__ f2)
{
  int idx = blockIdx.x * 256 + threadIdx.x;     // ((b*6+h)*2048 + n)
  if (idx >= Bq * Hh * Nn) return;
  int n  = idx & (Nn - 1);
  int bh = idx >> 11;
  int h  = bh % Hh;
  int b  = bh / Hh;
  const float* hp = Hmat + (b * Nn + n) * C96 + h * Oo;
  const float* A1 = a1 + h * Oo;
  const float* A2 = a2 + h * Oo;
  float s1 = b1[h], s2 = b2[h];
#pragma unroll
  for (int o = 0; o < Oo; o++) {
    float hv = hp[o];
    s1 = fmaf(hv, A1[o], s1);
    s2 = fmaf(hv, A2[o], s2);
  }
  f1[idx] = s1;
  f2[idx] = s2;
}

// ---------------- Kernel 5: fused GAT attention (flash-style softmax, bf16 WMMA P@H)
// grid (N/16, H, B), one wave per block, 16x16 output tile.
__global__ __launch_bounds__(32) void k_attn(
    const float* __restrict__ f1a, const float* __restrict__ f2a,
    const float* __restrict__ adj, const __bf16* __restrict__ HbfT,
    const float* __restrict__ Resmat, const float* __restrict__ br,
    float* __restrict__ out_pre)
{
  int it = blockIdx.x, h = blockIdx.y, b = blockIdx.z;
  int i0 = it * 16;
  int l  = threadIdx.x;
  int lm = l & 15;
  bool hi = l >= 16;
  int bh  = b * Hh + h;

  float f1v = f1a[bh * Nn + i0 + lm];                 // this lane's row i = lm
  const float* f2p  = f2a + bh * Nn;
  const float* adjr = adj + (i0 + lm) * Nn;
  const __bf16* hbp = HbfT + (b * C96 + h * Oo + lm) * Nn + (hi ? 8 : 0);
  int off0 = hi ? 8 : 0;

  v8f acc = {};
  float rs = 0.0f;                                    // per-lane partial row-sum of exp

  // Warm the first ~1KB of each stream (global_prefetch_b8).
  __builtin_prefetch((const void*)(adjr), 0, 3);
  __builtin_prefetch((const void*)(f2p + off0), 0, 3);
  __builtin_prefetch((const void*)(hbp), 0, 3);

  for (int jt = 0; jt < Nn / 32; jt++) {
    int jb = jt * 32;

    // Stream-ahead prefetch ~8 k-tiles (1KB of adj row) into the cache hierarchy.
    if (jt + 8 < Nn / 32) {
      __builtin_prefetch((const void*)(adjr + jb + 256), 0, 3);
      __builtin_prefetch((const void*)(f2p + jb + 256 + off0), 0, 3);
      __builtin_prefetch((const void*)(hbp + jb + 256), 0, 3);
    }

    v4f f2lo0 = *(const v4f*)(f2p + jb + off0);
    v4f f2lo1 = *(const v4f*)(f2p + jb + off0 + 4);
    v4f f2hi0 = *(const v4f*)(f2p + jb + 16 + off0);
    v4f f2hi1 = *(const v4f*)(f2p + jb + 16 + off0 + 4);
    v4f adlo0 = *(const v4f*)(adjr + jb + off0);
    v4f adlo1 = *(const v4f*)(adjr + jb + off0 + 4);
    v4f adhi0 = *(const v4f*)(adjr + jb + 16 + off0);
    v4f adhi1 = *(const v4f*)(adjr + jb + 16 + off0 + 4);

    float f2v[16], adv[16];
#pragma unroll
    for (int i = 0; i < 4; i++) {
      f2v[i] = f2lo0[i]; f2v[i + 4] = f2lo1[i]; f2v[i + 8] = f2hi0[i]; f2v[i + 12] = f2hi1[i];
      adv[i] = adlo0[i]; adv[i + 4] = adlo1[i]; adv[i + 8] = adhi0[i]; adv[i + 12] = adhi1[i];
    }

    v16bf A;
#pragma unroll
    for (int i = 0; i < 16; i++) {
      float s = f1v + f2v[i];
      s = (s > 0.0f) ? s : 0.2f * s;                  // leaky_relu(0.2)
      s += adv[i];
      float p = __expf(s);                            // logits bounded -> no max-sub needed
      rs += p;
      A[i] = (__bf16)p;
    }

    v8bf b0 = *(const v8bf*)(hbp + jb);
    v8bf b1v = *(const v8bf*)(hbp + jb + 16);
    v16bf Bf;
#pragma unroll
    for (int i = 0; i < 8; i++) { Bf[i] = b0[i]; Bf[i + 8] = b1v[i]; }

    acc = __builtin_amdgcn_wmma_f32_16x16x32_bf16(false, A, false, Bf, (short)0, acc, false, false);
  }

  // row sum for row lm lives in lanes lm and lm+16
  float tot = rs + __shfl_xor(rs, 16);
  float inv = 1.0f / tot;

  int colg = h * Oo + lm;
#pragma unroll
  for (int r = 0; r < 8; r++) {
    int rowm = r + (hi ? 8 : 0);
    float sc = __shfl(inv, rowm);                     // lane rowm holds row rowm's 1/sum
    int n = i0 + rowm;
    float ret = acc[r] * sc + Resmat[(b * Nn + n) * C96 + colg] + br[colg];
    // relu(transpose(elu(ret))) == max(ret, 0)
    out_pre[(b * Nn + n) * C96 + colg] = fmaxf(ret, 0.0f);
  }
}

// ---------------- Kernel 6: BatchNorm stats per node channel n (reduce over B,H,O = 384)
__global__ __launch_bounds__(128) void k_bnstats(const float* __restrict__ out_pre,
                                                 float* __restrict__ mean,
                                                 float* __restrict__ invstd)
{
  int n = blockIdx.x;
  int t = threadIdx.x;
  float s = 0.0f, ss = 0.0f;
  for (int k = t; k < Bq * C96; k += 128) {
    int b = k / C96, r = k % C96;
    float v = out_pre[(b * Nn + n) * C96 + r];
    s += v; ss += v * v;
  }
  __shared__ float sh[128], sh2[128];
  sh[t] = s; sh2[t] = ss;
  __syncthreads();
  for (int st = 64; st > 0; st >>= 1) {
    if (t < st) { sh[t] += sh[t + st]; sh2[t] += sh2[t + st]; }
    __syncthreads();
  }
  if (t == 0) {
    float m = sh[0] * (1.0f / (Bq * C96));
    float v = sh2[0] * (1.0f / (Bq * C96)) - m * m;
    mean[n]   = m;
    invstd[n] = rsqrtf(v + 1e-5f);
  }
}

// ---------------- Kernel 7: BN apply -> output (B,N,H,O)
__global__ void k_bnapply(const float* __restrict__ out_pre,
                          const float* __restrict__ mean, const float* __restrict__ invstd,
                          const float* __restrict__ gamma, const float* __restrict__ beta,
                          float* __restrict__ out)
{
  int i = blockIdx.x * 256 + threadIdx.x;
  if (i >= ROWS * C96) return;
  int n = (i / C96) & (Nn - 1);
  out[i] = (out_pre[i] - mean[n]) * invstd[n] * gamma[n] + beta[n];
}

// ---------------- Host orchestration
extern "C" void kernel_launch(void* const* d_in, const int* in_sizes, int n_in,
                              void* d_out, int out_size, void* d_ws, size_t ws_size,
                              hipStream_t stream)
{
  const float* X    = (const float*)d_in[0];
  const float* adj  = (const float*)d_in[1];
  const float* Wt1  = (const float*)d_in[2];
  const float* bt1  = (const float*)d_in[3];
  const float* Wt2  = (const float*)d_in[4];
  const float* bt2  = (const float*)d_in[5];
  const float* Wt3  = (const float*)d_in[6];
  const float* bt3  = (const float*)d_in[7];
  const float* Wg   = (const float*)d_in[8];
  const float* a1   = (const float*)d_in[9];
  const float* b1   = (const float*)d_in[10];
  const float* a2   = (const float*)d_in[11];
  const float* b2   = (const float*)d_in[12];
  const float* Wr   = (const float*)d_in[13];
  const float* br   = (const float*)d_in[14];
  const float* gamma= (const float*)d_in[15];
  const float* beta = (const float*)d_in[16];
  float* out = (float*)d_out;

  char* w = (char*)d_ws;
  size_t off = 0;
  __bf16* Xbf   = (__bf16*)(w + off); off += (size_t)ROWS * Ff * 2;        // 10.5 MB
  __bf16* Wgbf  = (__bf16*)(w + off); off += (size_t)C96 * Ff * 2;
  __bf16* Wrbf  = (__bf16*)(w + off); off += (size_t)C96 * Ff * 2;
  float*  Hmat  = (float*)(w + off);  off += (size_t)ROWS * C96 * 4;       // 3.1 MB
  float*  Resm  = (float*)(w + off);  off += (size_t)ROWS * C96 * 4;
  __bf16* HbfT  = (__bf16*)(w + off); off += (size_t)ROWS * C96 * 2;
  float*  f1    = (float*)(w + off);  off += (size_t)Bq * Hh * Nn * 4;
  float*  f2    = (float*)(w + off);  off += (size_t)Bq * Hh * Nn * 4;
  float*  opre  = (float*)(w + off);  off += (size_t)ROWS * C96 * 4;
  float*  mean  = (float*)(w + off);  off += (size_t)Nn * 4;
  float*  istd  = (float*)(w + off);  off += (size_t)Nn * 4;

  k_timeblock<<<ROWS, 64, 0, stream>>>(X, Wt1, bt1, Wt2, bt2, Wt3, bt3, Xbf);
  k_pack<<<(C96 * Ff + 255) / 256, 256, 0, stream>>>(Wg, Wr, Wgbf, Wrbf);
  k_gemm<<<ROWS / 16, 32, 0, stream>>>(Xbf, Wgbf, Wrbf, Hmat, Resm, HbfT);
  k_f1f2<<<(Bq * Hh * Nn + 255) / 256, 256, 0, stream>>>(Hmat, a1, b1, a2, b2, f1, f2);
  k_attn<<<dim3(Nn / 16, Hh, Bq), 32, 0, stream>>>(f1, f2, adj, HbfT, Resm, br, opre);
  k_bnstats<<<Nn, 128, 0, stream>>>(opre, mean, istd);
  k_bnapply<<<(ROWS * C96 + 255) / 256, 256, 0, stream>>>(opre, mean, istd, gamma, beta, out);
}
